// VGG_48052094107905
// MI455X (gfx1250) — matrix operand order repaired and checked by reading, not compiled, not run on hardware
//
#include <hip/hip_runtime.h>
#include <math.h>

// ---------------------------------------------------------------------------
// CDNA5 (gfx1250) deformable-transposed-conv VGG.
// - Per-tap GEMMs: v_wmma_f32_16x16x32_f16, all 9 taps accumulated in one wave
//   (9 x v8f accumulators) so each A fragment feeds 9 WMMAs.
// - Weights staged through LDS with global_load_async_to_lds_b128 (ASYNCcnt),
//   shared by 4 waves per workgroup -> 4x fewer L2 weight reads.
// - Bilinear scatter fused into the WMMA epilogue via global f32 atomics.
// - Offset/mask convs stay fp32 (they produce coordinates). Conv biases + b1
//   are elided (exactly cancelled by train-mode BatchNorm).
// ---------------------------------------------------------------------------

typedef __attribute__((ext_vector_type(16))) _Float16 v16h;
typedef __attribute__((ext_vector_type(8)))  _Float16 v8h;
typedef __attribute__((ext_vector_type(8)))  float    v8f;
typedef __attribute__((ext_vector_type(4)))  int      v4i;

#if defined(__HIP_DEVICE_COMPILE__) && \
    __has_builtin(__builtin_amdgcn_global_load_async_to_lds_b128) && \
    __has_builtin(__builtin_amdgcn_s_wait_asynccnt)
#define ASYNC_LDS 1
// exact parameter types per hipcc diagnostic: v4i in AS1 (global) / AS3 (LDS)
typedef __attribute__((address_space(1))) v4i* g_v4i_p;
typedef __attribute__((address_space(3))) v4i* l_v4i_p;
#else
#define ASYNC_LDS 0
#endif

static inline unsigned cdiv(unsigned a, unsigned b) { return (a + b - 1) / b; }

// NCHW fp32 -> pixel-major [B*H*W, C] fp32
__global__ void nchw_to_pc(const float* __restrict__ in, float* __restrict__ out,
                           int B, int C, int H, int W) {
  int idx = blockIdx.x * blockDim.x + threadIdx.x;
  int total = B * C * H * W;
  if (idx >= total) return;
  int c = idx % C;
  int p = idx / C;
  int b = p / (H * W);
  int r = p % (H * W);
  out[idx] = in[(size_t)(b * C + c) * H * W + r];
}

// fp32 [N,C] -> f16 [N,Cp] zero-padded columns
__global__ void act_to_f16(const float* __restrict__ a, _Float16* __restrict__ o,
                           int N, int C, int Cp) {
  int idx = blockIdx.x * blockDim.x + threadIdx.x;
  if (idx >= N * Cp) return;
  int c = idx % Cp;
  int r = idx / Cp;
  o[idx] = (c < C) ? (_Float16)a[(size_t)r * C + c] : (_Float16)0.f;
}

// ConvTranspose weight [Cin][Cout][3][3] fp32 -> B^T tiles [9][Cout][Cinp] f16
__global__ void conv_w_to_bT(const float* __restrict__ w, _Float16* __restrict__ wt,
                             int Cin, int Cout, int Cinp) {
  int idx = blockIdx.x * blockDim.x + threadIdx.x;
  int total = 9 * Cout * Cinp;
  if (idx >= total) return;
  int i = idx % Cinp;
  int t = idx / Cinp;
  int o = t % Cout;
  int k = t / Cout;
  wt[idx] = (i < Cin) ? (_Float16)w[((size_t)i * Cout + o) * 9 + k] : (_Float16)0.f;
}

// Dense matrix [K][Nout] fp32 -> B^T [Nout][K] f16  (for the FC1 head GEMM)
__global__ void mat_to_bT(const float* __restrict__ w, _Float16* __restrict__ wt,
                          int K, int Nout) {
  int idx = blockIdx.x * blockDim.x + threadIdx.x;
  if (idx >= K * Nout) return;
  int i = idx % K;
  int o = idx / K;
  wt[idx] = (_Float16)w[(size_t)i * Nout + o];
}

// Offset/mask 3x3 convs (fp32, pad=1). Produces ABSOLUTE fractional targets
// py,px per tap and sigmoided masks. o in [0,18) -> offsets, [18,27) -> masks.
__global__ void offmask_conv(const float* __restrict__ act,
                             const float* __restrict__ wo, const float* __restrict__ ob,
                             const float* __restrict__ wm, const float* __restrict__ mb,
                             float* __restrict__ POFF, float* __restrict__ PMASK,
                             int H, int W, int Cin, int N) {
  int idx = blockIdx.x * blockDim.x + threadIdx.x;
  if (idx >= N * 27) return;
  int o = idx % 27;
  int p = idx / 27;
  int HW = H * W;
  int b = p / HW;
  int r = p % HW;
  int y = r / W;
  int x = r % W;
  const float* wptr = (o < 18) ? (wo + (size_t)o * Cin * 9)
                               : (wm + (size_t)(o - 18) * Cin * 9);
  float acc = 0.f;
  for (int ty = 0; ty < 3; ++ty) {
    int yy = y + ty - 1;
    if (yy < 0 || yy >= H) continue;
    for (int tx = 0; tx < 3; ++tx) {
      int xx = x + tx - 1;
      if (xx < 0 || xx >= W) continue;
      const float* arow = act + ((size_t)(b * HW) + yy * W + xx) * Cin;
      const float* wrow = wptr + ty * 3 + tx;
      float s = 0.f;
      for (int ci = 0; ci < Cin; ++ci) s += wrow[(size_t)ci * 9] * arow[ci];
      acc += s;
    }
  }
  if (o < 18) {
    acc += ob[o];
    int k = o >> 1;
    float base = (o & 1) ? (float)(x + (k % 3) - 1) : (float)(y + (k / 3) - 1);
    POFF[((size_t)p * 9 + k) * 2 + (o & 1)] = acc + base;
  } else {
    int k = o - 18;
    acc += mb[k];
    PMASK[(size_t)p * 9 + k] = 1.f / (1.f + expf(-acc));
  }
}

// Fused all-tap WMMA GEMM + modulated bilinear scatter.
// Grid: (N/64, Cout/16); block = 4 waves. Each wave owns one 16-pixel tile of
// the same 16-channel block; the 9-tap weight fragments for a 128-wide K
// window (36 KB) are staged into LDS once per workgroup via async loads and
// consumed by all 4 waves (ds_load_b128 -> WMMA B operands).
__global__ __launch_bounds__(128)
void dconv_taps_wmma(const _Float16* __restrict__ X16,  // [N][Cinp]
                     const _Float16* __restrict__ W16,  // [9][Cout][Cinp]
                     const float* __restrict__ POFF,    // [N][9][2]
                     const float* __restrict__ PMASK,   // [N][9]
                     float* __restrict__ OUT,           // [N][Cout] accumulator
                     int Cinp, int Cout, int H, int W) {
  // 4 K-chunks x 9 taps x (16 cols x 32 K) f16 = 36 KB
  __shared__ __align__(32) _Float16 sB[4 * 9 * 512];

  const int tid   = threadIdx.x;
  const int lane  = tid & 31;
  const int wave  = tid >> 5;
  const int colT0 = blockIdx.y * 16;
  const int row   = (blockIdx.x * 4 + wave) * 16 + (lane & 15);  // A row (pixel)
  const int asel  = (lane & 16) ? 8 : 0;   // A: K {0-7,16-23} / {8-15,24-31}

  const _Float16* arow = X16 + (size_t)row * Cinp;

  v8f acc[9] = {};
  for (int kc0 = 0; kc0 < Cinp; kc0 += 128) {
    int rem = (Cinp - kc0) >> 5;
    const int chunks = rem < 4 ? rem : 4;
    __syncthreads();                       // previous stage fully consumed

    // Cooperative stage: fragment f = chunk*9+tap is 32 lane-slots x 32 B,
    // laid out so compute lane `ln` reads 32 contiguous bytes.
    const int nslot = chunks * 9 * 64;     // 16-byte slots
    for (int s = tid; s < nslot; s += 128) {
      const int f   = s >> 6;
      const int u   = s & 63;
      const int ln  = u >> 1;
      const int sel = u & 1;
      const int chunk = f / 9, tap = f % 9;
      const _Float16* gsrc = W16 + ((size_t)tap * Cout + colT0 + (ln & 15)) * Cinp
                             + kc0 + chunk * 32 + ((ln & 16) ? 8 : 0) + sel * 16;
      _Float16* ldst = &sB[f * 512 + ln * 16 + sel * 8];
#if ASYNC_LDS
      __builtin_amdgcn_global_load_async_to_lds_b128((g_v4i_p)gsrc, (l_v4i_p)ldst,
                                                     0, 0);
#else
      *(v8h*)ldst = *(const v8h*)gsrc;
#endif
    }
#if ASYNC_LDS
    __builtin_amdgcn_s_wait_asynccnt(0);
#endif
    __syncthreads();

    for (int chunk = 0; chunk < chunks; ++chunk) {
      const int kc = kc0 + chunk * 32;
      v16h a;
      ((v8h*)&a)[0] = *(const v8h*)(arow + kc + asel);
      ((v8h*)&a)[1] = *(const v8h*)(arow + kc + asel + 16);
      __builtin_prefetch(arow + kc + 32, 0, 1);   // stream A
#pragma unroll
      for (int tap = 0; tap < 9; ++tap) {
        v16h b = *(const v16h*)&sB[(chunk * 9 + tap) * 512 + lane * 16];
        acc[tap] = __builtin_amdgcn_wmma_f32_16x16x32_f16(false, a, false, b,
                                                          (short)0, acc[tap],
                                                          false, false);
      }
    }
  }

  // Scatter epilogue: lane holds channel cc for 8 pixels m0..m0+7, 9 taps.
  // Fully unrolled so acc[] stays in VGPRs (no scratch).
  const int HW = H * W;
  const int cc = colT0 + (lane & 15);
  const int m0 = (blockIdx.x * 4 + wave) * 16 + ((lane & 16) ? 8 : 0);
#pragma unroll
  for (int k = 0; k < 9; ++k) {
#pragma unroll
    for (int r = 0; r < 8; ++r) {
      int p = m0 + r;
      float val = acc[k][r] * PMASK[(size_t)p * 9 + k];
      float py = POFF[((size_t)p * 9 + k) * 2 + 0];
      float px = POFF[((size_t)p * 9 + k) * 2 + 1];
      int b0 = p / HW;
      float y0 = floorf(py), x0 = floorf(px);
#pragma unroll
      for (int dy = 0; dy < 2; ++dy) {
#pragma unroll
        for (int dx = 0; dx < 2; ++dx) {
          float iy = y0 + (float)dy, ix = x0 + (float)dx;
          if (iy < 0.f || iy >= (float)H || ix < 0.f || ix >= (float)W) continue;
          float wgt = (1.f - fabsf(py - iy)) * (1.f - fabsf(px - ix));
          int q = b0 * HW + (int)iy * W + (int)ix;
          atomicAdd(&OUT[(size_t)q * Cout + cc], val * wgt);
        }
      }
    }
  }
}

// Plain WMMA GEMM C[M,Nout] = A[M,K] * B (B given transposed [Nout][K]).
__global__ __launch_bounds__(32)
void gemm_wmma_bT(const _Float16* __restrict__ A, const _Float16* __restrict__ BT,
                  float* __restrict__ C, int K, int Nout) {
  const int lane = threadIdx.x;
  const int row  = blockIdx.x * 16 + (lane & 15);
  const int colT = blockIdx.y * 16 + (lane & 15);
  const int asel = (lane & 16) ? 16 : 0;
  const int bsel = (lane & 16) ? 16 : 0;
  const _Float16* arow = A + (size_t)row * K;
  const _Float16* brow = BT + (size_t)colT * K + bsel;
  v8f acc = {};
  for (int kc = 0; kc < K; kc += 32) {
    v16h a, b;
    a = *(const v16h*)(arow + kc + asel);
    b = *(const v16h*)(brow + kc);
    acc = __builtin_amdgcn_wmma_f32_16x16x32_f16(false, a, false, b,
                                                 (short)0, acc, false, false);
  }
  const int cc = blockIdx.y * 16 + (lane & 15);
  const int m0 = blockIdx.x * 16 + ((lane & 16) ? 8 : 0);
#pragma unroll
  for (int r = 0; r < 8; ++r)
    C[(size_t)(m0 + r) * Nout + cc] = acc[r];
}

// Per-channel batch statistics (biased variance) over [N,C] pixel-major data.
__global__ void bn_stats(const float* __restrict__ x, int N, int C,
                         float* __restrict__ mean, float* __restrict__ var) {
  __shared__ float s1[256], s2[256];
  int c = blockIdx.x;
  int tid = threadIdx.x;
  float a = 0.f, b = 0.f;
  for (int i = tid; i < N; i += blockDim.x) {
    float v = x[(size_t)i * C + c];
    a += v;
    b += v * v;
  }
  s1[tid] = a; s2[tid] = b;
  __syncthreads();
  for (int s = 128; s > 0; s >>= 1) {
    if (tid < s) { s1[tid] += s1[tid + s]; s2[tid] += s2[tid + s]; }
    __syncthreads();
  }
  if (tid == 0) {
    float m = s1[0] / (float)N;
    mean[c] = m;
    var[c]  = s2[0] / (float)N - m * m;
  }
}

__global__ void bn_apply_relu(const float* __restrict__ x,
                              const float* __restrict__ mean, const float* __restrict__ var,
                              const float* __restrict__ g, const float* __restrict__ bt,
                              float* __restrict__ out, int N, int C) {
  int idx = blockIdx.x * blockDim.x + threadIdx.x;
  if (idx >= N * C) return;
  int c = idx % C;
  float y = g[c] * (x[idx] - mean[c]) * rsqrtf(var[c] + 1e-5f) + bt[c];
  out[idx] = fmaxf(y, 0.f);
}

// 2x2 maxpool on NHWC-flat
__global__ void maxpool2(const float* __restrict__ in, float* __restrict__ out,
                         int B, int H, int W, int C) {
  int Ho = H / 2, Wo = W / 2;
  int idx = blockIdx.x * blockDim.x + threadIdx.x;
  if (idx >= B * Ho * Wo * C) return;
  int c = idx % C;
  int t = idx / C;
  int xo = t % Wo; t /= Wo;
  int yo = t % Ho;
  int b = t / Ho;
  size_t base = (((size_t)b * H + 2 * yo) * W + 2 * xo) * C + c;
  float m = in[base];
  m = fmaxf(m, in[base + C]);
  m = fmaxf(m, in[base + (size_t)W * C]);
  m = fmaxf(m, in[base + (size_t)W * C + C]);
  out[idx] = m;
}

// 2x2 avgpool on [B,2,2,C] -> [B,C]
__global__ void avgpool22(const float* __restrict__ in, float* __restrict__ out,
                          int B, int C) {
  int idx = blockIdx.x * blockDim.x + threadIdx.x;
  if (idx >= B * C) return;
  int c = idx % C;
  int b = idx / C;
  size_t base = (size_t)b * 4 * C + c;
  out[idx] = 0.25f * (in[base] + in[base + C] + in[base + 2 * C] + in[base + 3 * C]);
}

// Tiny final FC: out[b][j] = b2[j] + sum_i h[b][i] * w2[i][j]   (Nout=10)
__global__ void fc2_kernel(const float* __restrict__ h, const float* __restrict__ w2,
                           const float* __restrict__ b2, float* __restrict__ out,
                           int B, int K, int Nout) {
  int idx = blockIdx.x * blockDim.x + threadIdx.x;
  if (idx >= B * Nout) return;
  int j = idx % Nout;
  int b = idx / Nout;
  float acc = b2[j];
  const float* hr = h + (size_t)b * K;
  for (int i = 0; i < K; ++i) acc += hr[i] * w2[(size_t)i * Nout + j];
  out[idx] = acc;
}

// ---------------------------------------------------------------------------

extern "C" void kernel_launch(void* const* d_in, const int* in_sizes, int n_in,
                              void* d_out, int out_size, void* d_ws, size_t ws_size,
                              hipStream_t stream) {
  (void)in_sizes; (void)out_size;
  if (n_in < 111) return;

  const int B = 128;
  const float* x = (const float*)d_in[0];
  const float *cw[13], *ow[13], *ob[13], *mw[13], *mb[13], *bg[13], *bb[13];
  for (int i = 0; i < 13; ++i) {
    cw[i] = (const float*)d_in[1 + i];     // conv_weights  [Cin,Cout,3,3]
    // d_in[14+i] conv_biases: elided (cancelled by BN)
    ow[i] = (const float*)d_in[27 + i];    // off_weights   [18,Cin,3,3]
    ob[i] = (const float*)d_in[40 + i];
    mw[i] = (const float*)d_in[53 + i];    // mask_weights  [9,Cin,3,3]
    mb[i] = (const float*)d_in[66 + i];
    bg[i] = (const float*)d_in[79 + i];
    bb[i] = (const float*)d_in[92 + i];
  }
  const float* w1  = (const float*)d_in[105];
  // d_in[106] b1: elided (cancelled by BN1d)
  const float* g1  = (const float*)d_in[107];
  const float* be1 = (const float*)d_in[108];
  const float* w2  = (const float*)d_in[109];
  const float* b2  = (const float*)d_in[110];

  // ---- workspace carve (working set ~137 MB; fits the 192 MB L2 regime) ----
  uint8_t* base = (uint8_t*)d_ws;
  size_t off = 0;
  auto take = [&](size_t bytes) -> void* {
    void* p = base + off;
    off = (off + bytes + 255) & ~(size_t)255;
    return p;
  };
  const size_t ACTB  = (size_t)131072 * 64 * 4;   // max N*C fp32
  float*    ACT0  = (float*)take(ACTB);
  float*    ACT1  = (float*)take(ACTB);
  float*    OUT   = (float*)take(ACTB);
  _Float16* X16   = (_Float16*)take((size_t)131072 * 64 * 2);
  _Float16* W16   = (_Float16*)take((size_t)9 * 512 * 512 * 2);
  float*    POFF  = (float*)take((size_t)131072 * 18 * 4);
  float*    PMASK = (float*)take((size_t)131072 * 9 * 4);
  float*    MEAN  = (float*)take(512 * 4);
  float*    VAR   = (float*)take(512 * 4);
  float*    HEADH = (float*)take((size_t)128 * 512 * 4);
  _Float16* HX16  = (_Float16*)take((size_t)128 * 512 * 2);
  _Float16* HW16  = (_Float16*)take((size_t)512 * 512 * 2);
  if (off > ws_size) return;

  struct LayerCfg { int Cin, Cout, H, pool; };
  const LayerCfg L[13] = {
    {3, 64, 32, 0},  {64, 64, 32, 1},
    {64, 128, 16, 0},{128, 128, 16, 1},
    {128, 256, 8, 0},{256, 256, 8, 0},{256, 256, 8, 1},
    {256, 512, 4, 0},{512, 512, 4, 0},{512, 512, 4, 1},
    {512, 512, 2, 0},{512, 512, 2, 0},{512, 512, 2, 0}};

  // Input NCHW -> pixel-major
  {
    int total = B * 3 * 32 * 32;
    nchw_to_pc<<<cdiv(total, 256), 256, 0, stream>>>(x, ACT0, B, 3, 32, 32);
  }

  float* cur = ACT0;
  float* nxt = ACT1;
  for (int l = 0; l < 13; ++l) {
    const int Cin = L[l].Cin, Cout = L[l].Cout, H = L[l].H, W = L[l].H;
    const int N = B * H * W;
    const int Cinp = (Cin + 31) & ~31;

    act_to_f16<<<cdiv(N * Cinp, 256), 256, 0, stream>>>(cur, X16, N, Cin, Cinp);
    conv_w_to_bT<<<cdiv(9 * Cout * Cinp, 256), 256, 0, stream>>>(cw[l], W16, Cin, Cout, Cinp);
    offmask_conv<<<cdiv(N * 27, 128), 128, 0, stream>>>(cur, ow[l], ob[l], mw[l], mb[l],
                                                        POFF, PMASK, H, W, Cin, N);
    (void)hipMemsetAsync(OUT, 0, (size_t)N * Cout * 4, stream);
    dim3 g(N / 64, Cout / 16);
    dconv_taps_wmma<<<g, 128, 0, stream>>>(X16, W16, POFF, PMASK, OUT, Cinp, Cout, H, W);
    bn_stats<<<Cout, 256, 0, stream>>>(OUT, N, Cout, MEAN, VAR);
    bn_apply_relu<<<cdiv(N * Cout, 256), 256, 0, stream>>>(OUT, MEAN, VAR, bg[l], bb[l],
                                                           nxt, N, Cout);
    if (L[l].pool) {
      int total = B * (H / 2) * (W / 2) * Cout;
      maxpool2<<<cdiv(total, 256), 256, 0, stream>>>(nxt, cur, B, H, W, Cout);
      // pooled result is the next layer's input in `cur`; no swap
    } else {
      float* t = cur; cur = nxt; nxt = t;
    }
  }

  // cur: [B,2,2,512] NHWC-flat -> avgpool -> [B,512]
  avgpool22<<<cdiv(B * 512, 256), 256, 0, stream>>>(cur, nxt, B, 512);

  // Head: relu(BN1d(X @ w1)) @ w2 + b2   (b1 cancelled by BN1d)
  act_to_f16<<<cdiv(128 * 512, 256), 256, 0, stream>>>(nxt, HX16, 128, 512, 512);
  mat_to_bT<<<cdiv(512 * 512, 256), 256, 0, stream>>>(w1, HW16, 512, 512);
  {
    dim3 g(128 / 16, 512 / 16);
    gemm_wmma_bT<<<g, 32, 0, stream>>>(HX16, HW16, OUT, 512, 512);
  }
  bn_stats<<<512, 256, 0, stream>>>(OUT, 128, 512, MEAN, VAR);
  bn_apply_relu<<<cdiv(128 * 512, 256), 256, 0, stream>>>(OUT, MEAN, VAR, g1, be1,
                                                          HEADH, 128, 512);
  fc2_kernel<<<cdiv(128 * 10, 128), 128, 0, stream>>>(HEADH, w2, b2, (float*)d_out,
                                                      128, 512, 10);
}
